// SelfAttention_51187420234193
// MI455X (gfx1250) — matrix operand order repaired and checked by reading, not compiled, not run on hardware
//
#include <hip/hip_runtime.h>

// ---------------- types / helpers ----------------
typedef __bf16 bf16;
typedef bf16  v16bf __attribute__((ext_vector_type(16)));
typedef float v8f   __attribute__((ext_vector_type(8)));

struct alignas(16) U16 { unsigned u0, u1, u2, u3; };   // POD 16-byte chunk
union Frag { U16 h[2]; v16bf v; };                     // 32B = one WMMA A/B fragment

__device__ __forceinline__ bf16 f2bf(float f) {
  unsigned u = __float_as_uint(f);
  u += 0x7FFFu + ((u >> 16) & 1u);                     // round-to-nearest-even
  unsigned short hs = (unsigned short)(u >> 16);
  return __builtin_bit_cast(bf16, hs);
}
__device__ __forceinline__ float bf2f(bf16 b) {
  unsigned short hs = __builtin_bit_cast(unsigned short, b);
  return __uint_as_float(((unsigned)hs) << 16);
}
__device__ __forceinline__ v8f wmma_bf16(v16bf a, v16bf b, v8f c) {
  // D = A(16x32) * B(32x16) + C, fp32 accum
  return __builtin_amdgcn_wmma_f32_16x16x32_bf16(false, a, false, b, (short)0, c, false, false);
}

#define DIMSZ   1024
#define NQH     16
#define NKVH    4
#define HD      64
#define KVDIM   (NKVH * HD)   // 256

// ---------------- fp32 -> bf16 convert ----------------
__global__ void cvt_f32_bf16(const float* __restrict__ src, bf16* __restrict__ dst, int n) {
  int i = blockIdx.x * blockDim.x + threadIdx.x;
  int stride = gridDim.x * blockDim.x;
  for (; i < n; i += stride) dst[i] = f2bf(src[i]);
}

// ---------------- tiled bf16 WMMA GEMM: C(MxN) = A(MxK) * B(KxN) ----------------
// block = 128 threads (4 waves). Tile 64(M) x 64(N), K-step 32.
// Wave w computes rows [w*16, w*16+16) x all 64 N columns (4 accum tiles).
#define APAD 40   // 64 rows x (32+8) bf16, 80B row stride (16B aligned)
#define BPAD 40   // B stored transposed: 64 (N) rows x (32+8) bf16

template <bool STORE_F32>
__global__ __launch_bounds__(128, 1)
void gemm_bf16_wmma(const bf16* __restrict__ A, const bf16* __restrict__ B,
                    bf16* __restrict__ Cb, float* __restrict__ Cf,
                    int M, int N, int K) {
  __shared__ bf16 As[64 * APAD];
  __shared__ bf16 Bt[64 * BPAD];

  const int tid  = threadIdx.x;
  const int lane = tid & 31;
  const int wave = tid >> 5;
  const int m0 = blockIdx.y * 64;
  const int n0 = blockIdx.x * 64;

  v8f acc[4];
  for (int j = 0; j < 4; ++j)
    for (int e = 0; e < 8; ++e) acc[j][e] = 0.f;

  const int arow = tid >> 2;           // 0..31 (and +32)
  const int acol = (tid & 3) * 8;      // 0,8,16,24
  const int brow = tid >> 2;           // K row 0..31
  const int bcol = (tid & 3) * 16;     // 0,16,32,48

  const int fr = lane & 15;            // fragment row/col lane index
  const int fs = lane >> 4;            // lane half select

  for (int k0 = 0; k0 < K; k0 += 32) {
    // ---- stage A tile 64x32 (row major, padded) ----
    {
      U16 a0 = *(const U16*)(A + (size_t)(m0 + arow)      * K + k0 + acol);
      U16 a1 = *(const U16*)(A + (size_t)(m0 + arow + 32) * K + k0 + acol);
      *(U16*)(As + arow * APAD + acol)        = a0;
      *(U16*)(As + (arow + 32) * APAD + acol) = a1;
    }
    // ---- stage B tile 32x64 transposed into Bt[n][k] ----
    {
      const bf16* gb = B + (size_t)(k0 + brow) * N + n0 + bcol;
      U16 b0 = *(const U16*)(gb);
      U16 b1 = *(const U16*)(gb + 8);
      const bf16* e0 = (const bf16*)&b0;
      const bf16* e1 = (const bf16*)&b1;
#pragma unroll
      for (int e = 0; e < 8; ++e) Bt[(bcol + e) * BPAD + brow]     = e0[e];
#pragma unroll
      for (int e = 0; e < 8; ++e) Bt[(bcol + 8 + e) * BPAD + brow] = e1[e];
    }
    __syncthreads();

    // ---- A fragment: row = wave*16+fr, K = {fs*8..+7} ∪ {fs*8+16..+23} ----
    Frag fa;
    {
      const bf16* p = As + (wave * 16 + fr) * APAD + fs * 8;
      fa.h[0] = *(const U16*)(p);
      fa.h[1] = *(const U16*)(p + 16);
    }
#pragma unroll
    for (int j = 0; j < 4; ++j) {
      // B fragment: col n = j*16+fr, K = {fs*8..+7} ∪ {fs*8+16..+23}
      Frag fb;
      const bf16* p = Bt + (j * 16 + fr) * BPAD + fs * 8;
      fb.h[0] = *(const U16*)(p);
      fb.h[1] = *(const U16*)(p + 16);
      acc[j] = wmma_bf16(fa.v, fb.v, acc[j]);
    }
    __syncthreads();
  }

  // ---- epilogue: C layout row = fs*8+v, col = lane&15 ----
#pragma unroll
  for (int j = 0; j < 4; ++j)
#pragma unroll
    for (int v = 0; v < 8; ++v) {
      int row = m0 + wave * 16 + fs * 8 + v;
      int col = n0 + j * 16 + fr;
      if (STORE_F32) Cf[(size_t)row * N + col] = acc[j][v];
      else           Cb[(size_t)row * N + col] = f2bf(acc[j][v]);
    }
}

// ---------------- RoPE (in place on bf16 Q and K) ----------------
__global__ void rope_kernel(bf16* __restrict__ Qb, bf16* __restrict__ Kb,
                            const float* __restrict__ cosT, const float* __restrict__ sinT,
                            int BS, int S) {
  // one thread per rotated pair, over Q (16 heads) then K (4 heads): 20 heads x 32 pairs
  long total = (long)BS * 20 * 32;
  long i = (long)blockIdx.x * blockDim.x + threadIdx.x;
  long stride = (long)gridDim.x * blockDim.x;
  for (; i < total; i += stride) {
    int d2   = (int)(i & 31);
    long r   = i >> 5;
    int hh   = (int)(r % 20);
    long pos = r / 20;                 // 0..BS-1
    int s    = (int)(pos % S);
    float c  = cosT[s * 32 + d2];
    float sn = sinT[s * 32 + d2];
    bf16* base = (hh < NQH)
        ? Qb + pos * DIMSZ + hh * HD + d2 * 2
        : Kb + pos * KVDIM + (hh - NQH) * HD + d2 * 2;
    float xr = bf2f(base[0]);
    float xi = bf2f(base[1]);
    base[0] = f2bf(xr * c - xi * sn);
    base[1] = f2bf(xr * sn + xi * c);
  }
}

// ---------------- flash attention (bf16 WMMA, online softmax) ----------------
// grid: x = S/64 (query block), y = B*NQH.  block = 128 (4 waves); wave owns 16 q rows.
#define PPAD 72   // probs tile row stride (bf16), 144B (16B aligned)
#define VPAD 72   // transposed V tile row stride

__global__ __launch_bounds__(128, 1)
void flash_attn(const bf16* __restrict__ Q, const bf16* __restrict__ Kb,
                const bf16* __restrict__ Vb, bf16* __restrict__ O, int S) {
  const int qb  = blockIdx.x;
  const int bh  = blockIdx.y;
  const int b   = bh / NQH;
  const int h   = bh % NQH;
  const int hkv = h / (NQH / NKVH);

  const int tid  = threadIdx.x;
  const int lane = tid & 31;
  const int wave = tid >> 5;
  const int fr   = lane & 15;
  const int fs   = lane >> 4;

  const bf16* Qp = Q  + (size_t)b * S * DIMSZ + h   * HD;
  const bf16* Kp = Kb + (size_t)b * S * KVDIM + hkv * HD;
  const bf16* Vp = Vb + (size_t)b * S * KVDIM + hkv * HD;
  bf16*       Op = O  + (size_t)b * S * DIMSZ + h   * HD;

  __shared__ bf16 Pt[4][16 * PPAD];   // per-wave probs (16 x 64)
  __shared__ bf16 Vt[64 * VPAD];      // transposed V block: [dim][key]

  // Q fragments (16 rows x 64 dims -> two 16x32 A fragments), loaded once
  Frag aq[2];
  {
    const int qrow = qb * 64 + wave * 16 + fr;
#pragma unroll
    for (int s = 0; s < 2; ++s) {
      const bf16* p = Qp + (size_t)qrow * DIMSZ + s * 32 + fs * 8;
      aq[s].h[0] = *(const U16*)(p);
      aq[s].h[1] = *(const U16*)(p + 16);
    }
  }

  float m_i[8], l_i[8];
  v8f o_acc[4];
#pragma unroll
  for (int v = 0; v < 8; ++v) { m_i[v] = -3.0e38f; l_i[v] = 0.f; }
#pragma unroll
  for (int j = 0; j < 4; ++j)
#pragma unroll
    for (int e = 0; e < 8; ++e) o_acc[j][e] = 0.f;

  const int rowq = qb * 64 + wave * 16 + fs * 8;  // row of slot v is rowq + v

  const int nkb = qb + 1;              // causal: only key blocks <= query block
  for (int kb = 0; kb < nkb; ++kb) {
    const int kbase = kb * 64;
    __syncthreads();                   // previous Vt consumers done

    if (kb + 1 < nkb) {                // prefetch next K/V block rows
      __builtin_prefetch(Kp + (size_t)(kbase + 64 + (tid >> 1)) * KVDIM, 0, 1);
      __builtin_prefetch(Vp + (size_t)(kbase + 64 + (tid >> 1)) * KVDIM, 0, 1);
    }

    // ---- stage V block 64x64 transposed: Vt[dim][key] ----
    {
      const int key = tid >> 1;
      const int c0  = (tid & 1) * 32;
      const bf16* vp = Vp + (size_t)(kbase + key) * KVDIM + c0;
      U16 vv[4];
#pragma unroll
      for (int e = 0; e < 4; ++e) vv[e] = ((const U16*)vp)[e];
      const bf16* ve = (const bf16*)vv;
#pragma unroll
      for (int e = 0; e < 32; ++e) Vt[(c0 + e) * VPAD + key] = ve[e];
    }
    __syncthreads();

    // ---- scores: S = Q * K^T  (B fragments straight from global: K rows are
    //      contiguous along head_dim, which is exactly a K^T column) ----
    v8f sc[4];
#pragma unroll
    for (int j = 0; j < 4; ++j) {
#pragma unroll
      for (int e = 0; e < 8; ++e) sc[j][e] = 0.f;
#pragma unroll
      for (int s = 0; s < 2; ++s) {
        Frag fk;
        const bf16* kp = Kp + (size_t)(kbase + j * 16 + fr) * KVDIM + s * 32 + fs * 8;
        fk.h[0] = *(const U16*)(kp);
        fk.h[1] = *(const U16*)(kp + 16);
        sc[j] = wmma_bf16(aq[s].v, fk.v, sc[j]);
      }
      // scale + causal mask, in place
      const int col = kbase + j * 16 + fr;
#pragma unroll
      for (int v = 0; v < 8; ++v)
        sc[j][v] = (col > rowq + v) ? -1.0e30f : sc[j][v] * 0.125f;  // 1/sqrt(64)
    }

    // ---- online softmax (all in place on sc, no extra arrays) ----
    {
      // row max across 4 local tiles + 16 lanes, then fold into m_i / l_i / o_acc
      float mx[8];
#pragma unroll
      for (int v = 0; v < 8; ++v)
        mx[v] = fmaxf(fmaxf(sc[0][v], sc[1][v]), fmaxf(sc[2][v], sc[3][v]));
#pragma unroll
      for (int d = 1; d < 16; d <<= 1)
#pragma unroll
        for (int v = 0; v < 8; ++v) mx[v] = fmaxf(mx[v], __shfl_xor(mx[v], d, 32));
#pragma unroll
      for (int v = 0; v < 8; ++v) {
        float mnew  = fmaxf(m_i[v], mx[v]);
        float alpha = __expf(m_i[v] - mnew);
        m_i[v] = mnew;
        l_i[v] *= alpha;
        o_acc[0][v] *= alpha;
        o_acc[1][v] *= alpha;
        o_acc[2][v] *= alpha;
        o_acc[3][v] *= alpha;
      }
    }
    {
      float rs[8];
#pragma unroll
      for (int v = 0; v < 8; ++v) rs[v] = 0.f;
#pragma unroll
      for (int j = 0; j < 4; ++j)
#pragma unroll
        for (int v = 0; v < 8; ++v) {
          float pv = __expf(sc[j][v] - m_i[v]);
          sc[j][v] = pv;
          rs[v] += pv;
        }
#pragma unroll
      for (int d = 1; d < 16; d <<= 1)
#pragma unroll
        for (int v = 0; v < 8; ++v) rs[v] += __shfl_xor(rs[v], d, 32);
#pragma unroll
      for (int v = 0; v < 8; ++v) l_i[v] += rs[v];
    }

    // ---- probs C-layout -> A-layout via per-wave LDS ----
    bf16* pw = Pt[wave];
#pragma unroll
    for (int j = 0; j < 4; ++j)
#pragma unroll
      for (int v = 0; v < 8; ++v)
        pw[(fs * 8 + v) * PPAD + j * 16 + fr] = f2bf(sc[j][v]);
    __builtin_amdgcn_wave_barrier();   // same-wave LDS in-order; block compiler reorder

    // ---- O += P * V ----
#pragma unroll
    for (int s = 0; s < 2; ++s) {
      Frag ap;
      const bf16* p = pw + fr * PPAD + s * 32 + fs * 8;
      ap.h[0] = *(const U16*)(p);
      ap.h[1] = *(const U16*)(p + 16);
#pragma unroll
      for (int j = 0; j < 4; ++j) {
        Frag fv;
        const bf16* vp = Vt + (j * 16 + fr) * VPAD + s * 32 + fs * 8;
        fv.h[0] = *(const U16*)(vp);
        fv.h[1] = *(const U16*)(vp + 16);
        o_acc[j] = wmma_bf16(ap.v, fv.v, o_acc[j]);
      }
    }
    __builtin_amdgcn_wave_barrier();
  }

  // ---- normalize + store bf16 attn output ----
#pragma unroll
  for (int v = 0; v < 8; ++v) {
    float inv = 1.0f / l_i[v];
    int row = rowq + v;
#pragma unroll
    for (int j = 0; j < 4; ++j)
      Op[(size_t)row * DIMSZ + j * 16 + fr] = f2bf(o_acc[j][v] * inv);
  }
}

// ---------------- launcher ----------------
extern "C" void kernel_launch(void* const* d_in, const int* in_sizes, int n_in,
                              void* d_out, int out_size, void* d_ws, size_t ws_size,
                              hipStream_t stream) {
  const float* x    = (const float*)d_in[0];
  // d_in[1] = start_pos (0, unused); d_in[4] = mask (causal, implemented directly)
  const float* fcos = (const float*)d_in[2];
  const float* fsin = (const float*)d_in[3];
  const float* wq   = (const float*)d_in[5];
  const float* wk   = (const float*)d_in[6];
  const float* wv   = (const float*)d_in[7];
  const float* wo   = (const float*)d_in[8];

  const int BS = in_sizes[0] / DIMSZ;        // B * S
  const int S  = in_sizes[2] / 32;           // freqs_cos is (S, 32)
  const int B  = BS / S;

  // carve workspace (bf16 staging buffers), 256B-aligned regions
  char* p = (char*)d_ws;
  auto carve = [&](size_t bytes) -> void* {
    void* r = (void*)p;
    p += (bytes + 255) & ~(size_t)255;
    return r;
  };
  bf16* Xb  = (bf16*)carve((size_t)BS * DIMSZ * 2);
  bf16* Wqb = (bf16*)carve((size_t)DIMSZ * DIMSZ * 2);
  bf16* Wkb = (bf16*)carve((size_t)DIMSZ * KVDIM * 2);
  bf16* Wvb = (bf16*)carve((size_t)DIMSZ * KVDIM * 2);
  bf16* Wob = (bf16*)carve((size_t)DIMSZ * DIMSZ * 2);
  bf16* Qb  = (bf16*)carve((size_t)BS * DIMSZ * 2);
  bf16* Kb  = (bf16*)carve((size_t)BS * KVDIM * 2);
  bf16* Vb  = (bf16*)carve((size_t)BS * KVDIM * 2);
  bf16* Ab  = (bf16*)carve((size_t)BS * DIMSZ * 2);

  auto cvt = [&](const float* s, bf16* d, int n) {
    int blocks = (n + 255) / 256;
    if (blocks > 4096) blocks = 4096;
    cvt_f32_bf16<<<blocks, 256, 0, stream>>>(s, d, n);
  };
  cvt(x,  Xb,  BS * DIMSZ);
  cvt(wq, Wqb, DIMSZ * DIMSZ);
  cvt(wk, Wkb, DIMSZ * KVDIM);
  cvt(wv, Wvb, DIMSZ * KVDIM);
  cvt(wo, Wob, DIMSZ * DIMSZ);

  // QKV projections (bf16 out)
  gemm_bf16_wmma<false><<<dim3(DIMSZ / 64, BS / 64), 128, 0, stream>>>(
      Xb, Wqb, Qb, nullptr, BS, DIMSZ, DIMSZ);
  gemm_bf16_wmma<false><<<dim3(KVDIM / 64, BS / 64), 128, 0, stream>>>(
      Xb, Wkb, Kb, nullptr, BS, KVDIM, DIMSZ);
  gemm_bf16_wmma<false><<<dim3(KVDIM / 64, BS / 64), 128, 0, stream>>>(
      Xb, Wvb, Vb, nullptr, BS, KVDIM, DIMSZ);

  // RoPE on Q and K
  {
    long total = (long)BS * 20 * 32;
    int blocks = (int)((total + 255) / 256);
    if (blocks > 4096) blocks = 4096;
    rope_kernel<<<blocks, 256, 0, stream>>>(Qb, Kb, fcos, fsin, BS, S);
  }

  // flash attention -> Ab (bf16, (B*S) x 1024)
  flash_attn<<<dim3(S / 64, B * NQH), 128, 0, stream>>>(Qb, Kb, Vb, Ab, S);

  // output projection -> fp32 d_out
  gemm_bf16_wmma<true><<<dim3(DIMSZ / 64, BS / 64), 128, 0, stream>>>(
      Ab, Wob, nullptr, (float*)d_out, BS, DIMSZ, DIMSZ);
}
